// MLAAttention_36189394437057
// MI455X (gfx1250) — compile-verified, matrix operand order checked by reading, not run on hardware
//
#include <hip/hip_runtime.h>

typedef __bf16 bf16_t;
typedef __attribute__((ext_vector_type(16))) __bf16 v16bf;
typedef __attribute__((ext_vector_type(8)))  __bf16 v8bf;
typedef __attribute__((ext_vector_type(8)))  float  v8f;

__device__ __forceinline__ bf16_t f2b(float f) {
    unsigned u = __builtin_bit_cast(unsigned, f);
    unsigned r = u + 0x7FFFu + ((u >> 16) & 1u);   // round-to-nearest-even
    unsigned short h = (unsigned short)(r >> 16);
    return __builtin_bit_cast(bf16_t, h);
}

__device__ __forceinline__ v16bf cat8(v8bf lo, v8bf hi) {
    return __builtin_shufflevector(lo, hi, 0,1,2,3,4,5,6,7,8,9,10,11,12,13,14,15);
}

__device__ __forceinline__ v8f wmma_bf16(v16bf a, v16bf b, v8f c) {
    return __builtin_amdgcn_wmma_f32_16x16x32_bf16(false, a, false, b, (short)0, c, false, false);
}

// ---------------------------------------------------------------------------
// fp32 [n] -> bf16 [n] straight conversion (activations).
// ---------------------------------------------------------------------------
__global__ __launch_bounds__(256) void to_bf16(
    const float* __restrict__ in, bf16_t* __restrict__ out, int n)
{
    const int i = (blockIdx.x * 256 + threadIdx.x) * 4;
    if (i < n) {
        float4 v = *(const float4*)(in + i);
        out[i+0] = f2b(v.x); out[i+1] = f2b(v.y);
        out[i+2] = f2b(v.z); out[i+3] = f2b(v.w);
    }
}

// ---------------------------------------------------------------------------
// fp32 W[K][N] -> bf16 Wt[N][K] (weights, transposed once so GEMM B-fragments
// are contiguous b128 loads). 32x32 tiles via LDS; K,N multiples of 32.
// ---------------------------------------------------------------------------
__global__ __launch_bounds__(256) void transpose_to_bf16(
    const float* __restrict__ W, bf16_t* __restrict__ Wt, int K, int N)
{
    __shared__ float t[32][33];
    const int n0 = blockIdx.x * 32, k0 = blockIdx.y * 32;
    const int tx = threadIdx.x & 31;
    const int ty = threadIdx.x >> 5;          // 8 rows per pass
    for (int i = 0; i < 32; i += 8)
        t[ty + i][tx] = W[(size_t)(k0 + ty + i) * N + n0 + tx];
    __syncthreads();
    for (int i = 0; i < 32; i += 8)
        Wt[(size_t)(n0 + ty + i) * K + k0 + tx] = f2b(t[tx][ty + i]);
}

// ---------------------------------------------------------------------------
// C[M,N] = A[M,K] @ Bt[N,K]^T, bf16 operands, f32 accumulate, LDS-free.
// Block tile 128x128: 8 waves (4x2), each wave 32x64 = 2x4 WMMA frags.
// All fragment loads are contiguous 16B (b128). out_bf16 selects C dtype.
// Requires M%128==0, K%32==0; N guarded (handles N=64).
// ---------------------------------------------------------------------------
__global__ __launch_bounds__(256) void gemm_bf16_tn(
    const bf16_t* __restrict__ A, const bf16_t* __restrict__ Bt,
    void* __restrict__ Cv, int M, int N, int K, int out_bf16)
{
    const int tid  = threadIdx.x;
    const int lane = tid & 31;
    const int wave = tid >> 5;
    const int wm   = wave >> 1;          // 0..3
    const int wn   = wave & 1;           // 0..1
    const int col  = lane & 15;
    const int hv   = lane >> 4;
    const int kb   = hv * 8;
    const int bm0  = blockIdx.y * 128;
    const int bn0  = blockIdx.x * 128;

    const bf16_t* pa[2];
    for (int i = 0; i < 2; i++)
        pa[i] = A + (size_t)(bm0 + wm*32 + i*16 + col) * K + kb;
    const bf16_t* pb[4];
    for (int j = 0; j < 4; j++) {
        int cn = bn0 + wn*64 + j*16 + col;
        if (cn >= N) cn = 0;             // clamp: load valid memory, store guarded
        pb[j] = Bt + (size_t)cn * K + kb;
    }

    v8f acc[2][4] = {};
#pragma unroll 2
    for (int k0 = 0; k0 < K; k0 += 32) {
        v16bf aq[2], bq[4];
        for (int i = 0; i < 2; i++)
            aq[i] = cat8(*(const v8bf*)(pa[i] + k0),
                         *(const v8bf*)(pa[i] + k0 + 16));
        for (int j = 0; j < 4; j++)
            bq[j] = cat8(*(const v8bf*)(pb[j] + k0),
                         *(const v8bf*)(pb[j] + k0 + 16));
        for (int i = 0; i < 2; i++)
            for (int j = 0; j < 4; j++)
                acc[i][j] = wmma_bf16(aq[i], bq[j], acc[i][j]);
    }

    for (int i = 0; i < 2; i++) {
        const int row0 = bm0 + wm*32 + i*16 + hv*8;   // C-layout: M = r + 8*hv
        for (int j = 0; j < 4; j++) {
            const int cn = bn0 + wn*64 + j*16 + col;
            if (cn < N) {
                if (out_bf16) {
                    bf16_t* C = (bf16_t*)Cv;
                    for (int r = 0; r < 8; r++)
                        C[(size_t)(row0 + r) * N + cn] = f2b(acc[i][j][r]);
                } else {
                    float* C = (float*)Cv;
                    for (int r = 0; r < 8; r++)
                        C[(size_t)(row0 + r) * N + cn] = acc[i][j][r];
                }
            }
        }
    }
}

// ---------------------------------------------------------------------------
// RMSNorm over rows of 256 (kv latent), bf16 out (feeds GEMM A operand).
// ---------------------------------------------------------------------------
__global__ __launch_bounds__(256) void rmsnorm256(
    const float* __restrict__ in, const float* __restrict__ g,
    bf16_t* __restrict__ out)
{
    const int row  = blockIdx.x * 8 + (threadIdx.x >> 5);
    const int lane = threadIdx.x & 31;
    const float* p = in + (size_t)row * 256;
    float v[8]; float ss = 0.f;
    for (int i = 0; i < 8; i++) { v[i] = p[i*32 + lane]; ss += v[i]*v[i]; }
    for (int o = 16; o > 0; o >>= 1) ss += __shfl_xor(ss, o, 32);
    const float rs = rsqrtf(ss * (1.f/256.f) + 1e-6f);
    bf16_t* q = out + (size_t)row * 256;
    for (int i = 0; i < 8; i++) q[i*32 + lane] = f2b(v[i] * rs * g[i*32 + lane]);
}

// ---------------------------------------------------------------------------
// Assemble Q: per-(b,s,h) rmsnorm(q_c,128)*g_q, rope(q_r,64); 1/sqrt(192)
// folded in. Output xq bf16 [b][h][s][192].
// ---------------------------------------------------------------------------
__global__ __launch_bounds__(256) void assemble_q(
    const float* __restrict__ qc, const float* __restrict__ qr,
    const float* __restrict__ cosb, const float* __restrict__ sinb,
    const float* __restrict__ gq, bf16_t* __restrict__ xq)
{
    const int w    = blockIdx.x * 8 + (threadIdx.x >> 5);  // 0..32767
    const int lane = threadIdx.x & 31;
    const int h    = w & 15;
    const int row  = w >> 4;          // b*S + s
    const int s    = row & 1023;
    const int b    = row >> 10;
    const float SC = 0.0721687836f;   // 1/sqrt(192)

    const float* pc = qc + (size_t)row * 2048 + h * 128;
    float v[4]; float ss = 0.f;
    for (int i = 0; i < 4; i++) { v[i] = pc[i*32 + lane]; ss += v[i]*v[i]; }
    for (int o = 16; o > 0; o >>= 1) ss += __shfl_xor(ss, o, 32);
    const float rs = rsqrtf(ss * (1.f/128.f) + 1e-6f);

    bf16_t* o = xq + (((size_t)(b*16 + h)) * 1024 + s) * 192;
    for (int i = 0; i < 4; i++) o[i*32 + lane] = f2b(v[i] * rs * gq[i*32 + lane] * SC);

    const float* pr = qr + (size_t)row * 1024 + h * 64;
    const float x0 = pr[lane], x1 = pr[lane + 32];
    const float* cs = cosb + (size_t)s * 64;
    const float* sn = sinb + (size_t)s * 64;
    o[128 + lane]      = f2b((x0 * cs[lane]      - x1 * sn[lane])      * SC);
    o[128 + 32 + lane] = f2b((x1 * cs[lane + 32] + x0 * sn[lane + 32]) * SC);
}

// ---------------------------------------------------------------------------
// Assemble K (k_c | rope(k_r) broadcast) -> xk bf16 [b][h][s][192],
// and V transposed -> xvt bf16 [b][h][d][s].
// ---------------------------------------------------------------------------
__global__ __launch_bounds__(256) void assemble_kv(
    const float* __restrict__ kc, const float* __restrict__ kr,
    const float* __restrict__ vf,
    const float* __restrict__ cosb, const float* __restrict__ sinb,
    bf16_t* __restrict__ xk, bf16_t* __restrict__ xvt)
{
    const int w    = blockIdx.x * 8 + (threadIdx.x >> 5);
    const int lane = threadIdx.x & 31;
    const int h    = w & 15;
    const int row  = w >> 4;
    const int s    = row & 1023;
    const int b    = row >> 10;
    const size_t bh = (size_t)(b*16 + h);

    bf16_t* ok = xk + (bh * 1024 + s) * 192;
    const float* pk = kc + (size_t)row * 2048 + h * 128;
    for (int i = 0; i < 4; i++) ok[i*32 + lane] = f2b(pk[i*32 + lane]);

    const float* pr = kr + (size_t)row * 64;
    const float x0 = pr[lane], x1 = pr[lane + 32];
    const float* cs = cosb + (size_t)s * 64;
    const float* sn = sinb + (size_t)s * 64;
    ok[128 + lane]      = f2b(x0 * cs[lane]      - x1 * sn[lane]);
    ok[128 + 32 + lane] = f2b(x1 * cs[lane + 32] + x0 * sn[lane + 32]);

    const float* pv = vf + (size_t)row * 2048 + h * 128;
    bf16_t* ov = xvt + bh * 128 * 1024 + s;
    for (int i = 0; i < 4; i++) ov[(size_t)(i*32 + lane) * 1024] = f2b(pv[i*32 + lane]);
}

// ---------------------------------------------------------------------------
// Causal flash attention. Block = 8 waves; each wave owns 16 queries x 192d.
// Key chunks of 32: 12 WMMAs QK^T + online softmax + 8 WMMAs PV.
// Output bf16 [b*S][NH*128] (feeds final GEMM directly).
// ---------------------------------------------------------------------------
__global__ __launch_bounds__(256) void mla_attn(
    const bf16_t* __restrict__ xq, const bf16_t* __restrict__ xk,
    const bf16_t* __restrict__ xvt, bf16_t* __restrict__ outp)
{
    __shared__ bf16_t P[8][16][40];   // per-wave 16x32 P-tile (padded)

    const int tid  = threadIdx.x;
    const int lane = tid & 31;
    const int wave = tid >> 5;
    const int col  = lane & 15;
    const int hv   = lane >> 4;
    const int kb   = hv * 8;

    const int blk   = blockIdx.x;     // 256 blocks
    const int qt    = blk & 7;
    const int bh    = blk >> 3;       // b*16 + h
    const int qbase = qt * 128 + wave * 16;

    const bf16_t* Q  = xq  + (size_t)bh * 1024 * 192;
    const bf16_t* Kk = xk  + (size_t)bh * 1024 * 192;
    const bf16_t* V  = xvt + (size_t)bh * 128 * 1024;

    v16bf qf[6];
    {
        const bf16_t* p = Q + (size_t)(qbase + col) * 192;
        for (int c = 0; c < 6; c++)
            qf[c] = cat8(*(const v8bf*)(p + c*32 + kb),
                         *(const v8bf*)(p + c*32 + kb + 16));
    }

    v8f acc[8] = {};
    float m[8], l[8];
    for (int r = 0; r < 8; r++) { m[r] = -1e30f; l[r] = 0.f; }

    const int nch = qt * 4 + 4;       // uniform per block (causal bound)
    for (int ck = 0; ck < nch; ck++) {
        const int k0 = ck * 32;

        v8f s0 = {}, s1 = {};
        for (int kc = 0; kc < 2; kc++) {
            const bf16_t* kp = Kk + (size_t)(k0 + kc*16 + col) * 192;
            v8f sc = {};
            for (int c = 0; c < 6; c++)
                sc = wmma_bf16(qf[c],
                               cat8(*(const v8bf*)(kp + c*32 + kb),
                                    *(const v8bf*)(kp + c*32 + kb + 16)),
                               sc);
            if (kc == 0) s0 = sc; else s1 = sc;
        }

        const int key0 = k0 + col;
        const int key1 = k0 + 16 + col;
        for (int r = 0; r < 8; r++) {
            const int qi = qbase + hv*8 + r;
            float a0 = (key0 <= qi) ? s0[r] : -1e30f;
            float a1 = (key1 <= qi) ? s1[r] : -1e30f;
            float mx = fmaxf(a0, a1);
            for (int o = 8; o > 0; o >>= 1) mx = fmaxf(mx, __shfl_xor(mx, o, 32));
            const float mn = fmaxf(m[r], mx);
            const float al = __expf(m[r] - mn);
            const float p0 = __expf(a0 - mn);
            const float p1 = __expf(a1 - mn);
            float sum = p0 + p1;
            for (int o = 8; o > 0; o >>= 1) sum += __shfl_xor(sum, o, 32);
            l[r] = l[r] * al + sum;
            m[r] = mn;
            for (int j = 0; j < 8; j++) acc[j][r] = acc[j][r] * al;
            P[wave][hv*8 + r][col]      = f2b(p0);
            P[wave][hv*8 + r][16 + col] = f2b(p1);
        }
        __syncthreads();

        v16bf pf = cat8(*(const v8bf*)&P[wave][col][kb],
                        *(const v8bf*)&P[wave][col][kb + 16]);
        __syncthreads();

        for (int j = 0; j < 8; j++) {
            const bf16_t* vp = V + (size_t)(j*16 + col) * 1024 + k0 + kb;
            acc[j] = wmma_bf16(pf,
                               cat8(*(const v8bf*)vp, *(const v8bf*)(vp + 16)),
                               acc[j]);
        }
    }

    const int q0 = qbase + hv * 8;
    const int h  = bh & 15, b = bh >> 4;
    for (int r = 0; r < 8; r++) {
        const float inv = 1.f / l[r];
        bf16_t* po = outp + ((size_t)(b*1024 + q0 + r)) * 2048 + h*128 + col;
        for (int j = 0; j < 8; j++) po[j*16] = f2b(acc[j][r] * inv);
    }
}

// ---------------------------------------------------------------------------
extern "C" void kernel_launch(void* const* d_in, const int* in_sizes, int n_in,
                              void* d_out, int out_size, void* d_ws, size_t ws_size,
                              hipStream_t stream)
{
    (void)in_sizes; (void)n_in; (void)out_size; (void)ws_size;
    const float* x       = (const float*)d_in[0];
    const float* cosb    = (const float*)d_in[1];
    const float* sinb    = (const float*)d_in[2];
    const float* w_qc    = (const float*)d_in[3];
    const float* w_qup   = (const float*)d_in[4];
    const float* w_qrope = (const float*)d_in[5];
    const float* w_kvc   = (const float*)d_in[6];
    const float* w_kupk  = (const float*)d_in[7];
    const float* w_kupv  = (const float*)d_in[8];
    const float* w_krope = (const float*)d_in[9];
    const float* w_o     = (const float*)d_in[10];
    const float* g_q     = (const float*)d_in[11];
    const float* g_kv    = (const float*)d_in[12];
    float* out = (float*)d_out;

    char* ws = (char*)d_ws;
    size_t off = 0;
    auto alloc = [&](size_t bytes) -> void* {
        void* p = ws + off;
        off += (bytes + 255) & ~(size_t)255;
        return p;
    };
    // bf16 transposed weights [N][K]
    bf16_t* wt_qc    = (bf16_t*)alloc(512ull  * 2048 * 2);
    bf16_t* wt_qup   = (bf16_t*)alloc(2048ull * 512  * 2);
    bf16_t* wt_qrope = (bf16_t*)alloc(1024ull * 512  * 2);
    bf16_t* wt_kvc   = (bf16_t*)alloc(256ull  * 2048 * 2);
    bf16_t* wt_kupk  = (bf16_t*)alloc(2048ull * 256  * 2);
    bf16_t* wt_kupv  = (bf16_t*)alloc(2048ull * 256  * 2);
    bf16_t* wt_krope = (bf16_t*)alloc(64ull   * 2048 * 2);
    bf16_t* wt_o     = (bf16_t*)alloc(2048ull * 2048 * 2);
    // activations
    bf16_t* xb       = (bf16_t*)alloc(2048ull * 2048 * 2);
    bf16_t* q_lat_b  = (bf16_t*)alloc(2048ull * 512  * 2);
    float*  kv_raw   = (float*) alloc(2048ull * 256  * 4);
    float*  kr_raw   = (float*) alloc(2048ull * 64   * 4);
    bf16_t* kv_lat_b = (bf16_t*)alloc(2048ull * 256  * 2);
    float*  qc_raw   = (float*) alloc(2048ull * 2048 * 4);
    float*  qr_raw   = (float*) alloc(2048ull * 1024 * 4);
    float*  kc_f32   = (float*) alloc(2048ull * 2048 * 4);
    float*  xv_f32   = (float*) alloc(2048ull * 2048 * 4);
    bf16_t* xq       = (bf16_t*)alloc(2ull * 16 * 1024 * 192 * 2);
    bf16_t* xk       = (bf16_t*)alloc(2ull * 16 * 1024 * 192 * 2);
    bf16_t* xvt      = (bf16_t*)alloc(2ull * 16 * 1024 * 128 * 2);
    bf16_t* attn_b   = (bf16_t*)alloc(2048ull * 2048 * 2);

    dim3 blk(256);
    auto tr = [&](const float* W, bf16_t* Wt, int K, int N) {
        dim3 g((unsigned)(N / 32), (unsigned)(K / 32));
        transpose_to_bf16<<<g, blk, 0, stream>>>(W, Wt, K, N);
    };
    auto gemm = [&](const bf16_t* A, const bf16_t* Bt, void* C,
                    int M, int N, int K, int obf) {
        dim3 g((unsigned)((N + 127) / 128), (unsigned)(M / 128));
        gemm_bf16_tn<<<g, blk, 0, stream>>>(A, Bt, C, M, N, K, obf);
    };

    // one-time bf16 conversion / weight transposes
    tr(w_qc,    wt_qc,    2048, 512);
    tr(w_qup,   wt_qup,   512,  2048);
    tr(w_qrope, wt_qrope, 512,  1024);
    tr(w_kvc,   wt_kvc,   2048, 256);
    tr(w_kupk,  wt_kupk,  256,  2048);
    tr(w_kupv,  wt_kupv,  256,  2048);
    tr(w_krope, wt_krope, 2048, 64);
    tr(w_o,     wt_o,     2048, 2048);
    to_bf16<<<4096, blk, 0, stream>>>(x, xb, 2048 * 2048);

    // projections
    gemm(xb, wt_qc,    q_lat_b, 2048, 512,  2048, 1);
    gemm(xb, wt_kvc,   kv_raw,  2048, 256,  2048, 0);
    gemm(xb, wt_krope, kr_raw,  2048, 64,   2048, 0);
    gemm(q_lat_b, wt_qup,   qc_raw, 2048, 2048, 512, 0);
    gemm(q_lat_b, wt_qrope, qr_raw, 2048, 1024, 512, 0);

    rmsnorm256<<<256, blk, 0, stream>>>(kv_raw, g_kv, kv_lat_b);

    gemm(kv_lat_b, wt_kupk, kc_f32, 2048, 2048, 256, 0);
    gemm(kv_lat_b, wt_kupv, xv_f32, 2048, 2048, 256, 0);

    assemble_q<<<4096, blk, 0, stream>>>(qc_raw, qr_raw, cosb, sinb, g_q, xq);
    assemble_kv<<<4096, blk, 0, stream>>>(kc_f32, kr_raw, xv_f32, cosb, sinb, xk, xvt);

    mla_attn<<<256, blk, 0, stream>>>(xq, xk, xvt, attn_b);

    gemm(attn_b, wt_o, out, 2048, 2048, 2048, 0);
}
